// CriterionAlignment_77996606095507
// MI455X (gfx1250) — compile-verified, matrix operand order checked by reading, not run on hardware
//
#include <hip/hip_runtime.h>
#include <cstdint>
#include <cstddef>

#define BATCH 256
#define TL 128
#define IL 127
#define DIM 1024
#define NITER 50

typedef __bf16 bf16;
typedef __attribute__((ext_vector_type(16))) __bf16 v16bf;
typedef __attribute__((ext_vector_type(8)))  float  v8f;
typedef __attribute__((ext_vector_type(4)))  unsigned int u32x4;
typedef __attribute__((ext_vector_type(8)))  int          i32x8;
typedef __attribute__((ext_vector_type(4)))  int          i32x4;

#if defined(__has_builtin)
#if __has_builtin(__builtin_amdgcn_tensor_load_to_lds)
#define HAVE_TDM 1
#endif
#endif
#ifndef HAVE_TDM
#define HAVE_TDM 0
#endif

// ---------------------------------------------------------------------------
// Kernel 1: per-row inverse L2 norm (one wave32 per row, 8 rows per block)
// ---------------------------------------------------------------------------
__global__ __launch_bounds__(256) void rownorm_kernel(
    const float* __restrict__ src, float* __restrict__ inv,
    int rowsPerBatch, int rowOffset, int batchStrideRows, int totalRows) {
  const int row  = blockIdx.x * 8 + (threadIdx.x >> 5);
  const int lane = threadIdx.x & 31;
  if (row >= totalRows) return;
  const int b = row / rowsPerBatch;
  const int r = row - b * rowsPerBatch;
  const float* p = src + ((size_t)b * batchStrideRows + rowOffset + r) * DIM;
  float s = 0.f;
  for (int i = lane; i < DIM; i += 32) { const float v = p[i]; s += v * v; }
  for (int off = 16; off > 0; off >>= 1) s += __shfl_xor(s, off, 32);
  if (lane == 0) inv[row] = 1.0f / fmaxf(sqrtf(s), 1e-5f);
}

// ---------------------------------------------------------------------------
// Kernel 2: cosine-cost batched GEMM via bf16 WMMA, fused mask + exp(-C/beta).
// One wave per 16x16 output tile; TWO independent accumulator chains break
// the WMMA->WMMA RAW dependency (5-slot hazard for bf16 per ISA 7.12.1).
// Per-lane operand layout per CDNA5 ISA 7.12.2 (16-bit A 16x32 / B 32x16).
// ---------------------------------------------------------------------------
__device__ __forceinline__ v16bf pack16(const float* __restrict__ p,
                                        int c0, int c1, float scale) {
  const float4 a0 = *reinterpret_cast<const float4*>(p + c0);
  const float4 a1 = *reinterpret_cast<const float4*>(p + c0 + 4);
  const float4 a2 = *reinterpret_cast<const float4*>(p + c1);
  const float4 a3 = *reinterpret_cast<const float4*>(p + c1 + 4);
  v16bf r;
  r[0]  = (bf16)(a0.x * scale); r[1]  = (bf16)(a0.y * scale);
  r[2]  = (bf16)(a0.z * scale); r[3]  = (bf16)(a0.w * scale);
  r[4]  = (bf16)(a1.x * scale); r[5]  = (bf16)(a1.y * scale);
  r[6]  = (bf16)(a1.z * scale); r[7]  = (bf16)(a1.w * scale);
  r[8]  = (bf16)(a2.x * scale); r[9]  = (bf16)(a2.y * scale);
  r[10] = (bf16)(a2.z * scale); r[11] = (bf16)(a2.w * scale);
  r[12] = (bf16)(a3.x * scale); r[13] = (bf16)(a3.y * scale);
  r[14] = (bf16)(a3.z * scale); r[15] = (bf16)(a3.w * scale);
  return r;
}

__global__ __launch_bounds__(32) void cosine_gemm_kernel(
    const float* __restrict__ X, const float* __restrict__ Yb,
    const float* __restrict__ invx, const float* __restrict__ invy,
    const int* __restrict__ tnum, const int* __restrict__ onum,
    float* __restrict__ Amat) {
  const int b    = blockIdx.y;
  const int mt   = blockIdx.x >> 3;
  const int nt   = blockIdx.x & 7;
  const int lane = threadIdx.x;
  const int half = lane >> 4;   // half selects K in [h8..h8+8) + [16+h8..), per ISA layout
  const int r16  = lane & 15;

  const int mrow  = mt * 16 + r16;                 // A-operand source row (X)
  const int nglob = nt * 16 + r16;                 // B-operand column (Y row)
  const int nrow  = (nglob < IL) ? nglob : (IL - 1);

  const float* xr = X  + ((size_t)b * TL + mrow) * DIM;
  const float* yr = Yb + (((size_t)b * 128) + 1 + nrow) * DIM;  // img = obj[:,1:]
  const float sx = invx[b * TL + mrow];
  const float sy = invy[b * IL + nrow];

  v8f acc0 = {};
  v8f acc1 = {};
#pragma unroll 4
  for (int k = 0; k < DIM; k += 64) {
    // chain 0: K block [k, k+32)
    {
      const int c0 = k + half * 8;
      const int c1 = k + 16 + half * 8;
      const v16bf av = pack16(xr, c0, c1, sx);
      const v16bf bv = pack16(yr, c0, c1, sy);
      acc0 = __builtin_amdgcn_wmma_f32_16x16x32_bf16(
          false, av, false, bv, (short)0, acc0, false, false);
    }
    // chain 1: K block [k+32, k+64)  (independent of chain 0)
    {
      const int c0 = k + 32 + half * 8;
      const int c1 = k + 48 + half * 8;
      const v16bf av = pack16(xr, c0, c1, sx);
      const v16bf bv = pack16(yr, c0, c1, sy);
      acc1 = __builtin_amdgcn_wmma_f32_16x16x32_bf16(
          false, av, false, bv, (short)0, acc1, false, false);
    }
  }

  bool npad = true;
  if (nglob < IL) npad = (onum[b * 128 + nglob + 1] == 0);
  float* Arow = Amat + ((size_t)b * 128 + nglob) * 128;
  const int mbase = mt * 16 + half * 8;            // D: M = r + 8*half
#pragma unroll
  for (int r = 0; r < 8; ++r) {
    const int m = mbase + r;
    const bool mpad = (tnum[b * TL + m] == 0);
    float val = 0.f;
    if (!npad && !mpad) {
      const float s = acc0[r] + acc1[r];
      const float cost = 1.0f - s;
      val = __expf(-2.0f * cost);                  // exp(-cost/beta), beta=0.5
    }
    Arow[m] = val;                                 // transposed store: A[n][m]
  }
}

// ---------------------------------------------------------------------------
// Kernel 3: IPOT iteration, one workgroup (128 threads) per batch element.
// A (127x128 f32, stride 129) loaded into LDS via Tensor Data Mover with
// HW padding (pad_interval=128 DW, pad_amount=1 DW -> stride 129); T in LDS.
// ---------------------------------------------------------------------------
__device__ __forceinline__ float block_reduce_sum(float v, float* red) {
  const int tid = threadIdx.x;
  red[tid] = v; __syncthreads();
  for (int off = 64; off > 0; off >>= 1) {
    if (tid < off) red[tid] += red[tid + off];
    __syncthreads();
  }
  const float r = red[0];
  __syncthreads();
  return r;
}

__global__ __launch_bounds__(128) void ipot_kernel(
    const float* __restrict__ Amat,
    const int* __restrict__ tnum, const int* __restrict__ onum,
    float* __restrict__ partials) {
  extern __shared__ float smem[];
  float* A_s     = smem;                  // 127*129
  float* T_s     = A_s + IL * 129;        // 127*129
  float* sigma_s = T_s + IL * 129;        // 128
  float* delta_s = sigma_s + 128;         // 128
  float* xmask   = delta_s + 128;         // 128
  float* ymask   = xmask + 128;           // 128
  float* red     = ymask + 128;           // 128

  const int b   = blockIdx.x;
  const int tid = threadIdx.x;

  const bool tp = (tnum[b * TL + tid] == 0);
  bool yp = true;
  if (tid < IL) yp = (onum[b * 128 + tid + 1] == 0);
  xmask[tid] = tp ? 10000.f : 0.f;
  ymask[tid] = yp ? 10000.f : 0.f;

  const float xl = block_reduce_sum(tp ? 0.f : 1.f, red);
  const float yl = block_reduce_sum((tid < IL && !yp) ? 1.f : 0.f, red);
  sigma_s[tid] = tp ? 0.f : (1.0f / xl);

  const float* Ag = Amat + (size_t)b * (128 * 128);
#if HAVE_TDM
  if (tid < 32) {   // one wave issues the TDM descriptor
    const unsigned long long ga = (unsigned long long)(uintptr_t)Ag;
    const unsigned lds = (unsigned)(uintptr_t)A_s;   // low 32b = LDS offset
    u32x4 g0 = {};
    g0[0] = 1u;                                       // count=1, user mode
    g0[1] = lds;                                      // lds_addr
    g0[2] = (unsigned)(ga & 0xffffffffull);           // global_addr[31:0]
    g0[3] = (unsigned)((ga >> 32) & 0x1ffffffull) | (2u << 30);  // addr[56:32]|type=2
    i32x8 g1 = {};
    g1[0] = (int)((2u << 16)      // data_size = 4B
                | (1u << 20)      // pad_enable
                | (6u << 22));    // pad_interval = 128 DW; pad_amount=0 -> 1 DW
    g1[1] = (int)(128u << 16);    // tensor_dim0 = 128   (bits 79:48)
    g1[2] = (int)(127u << 16);    // tensor_dim1 = 127   (bits 111:80)
    g1[3] = (int)(128u << 16);    // tile_dim0   = 128   (bits 127:112)
    g1[4] = (int)(127u);          // tile_dim1 = 127, tile_dim2 = 0
    g1[5] = (int)(128u);          // tensor_dim0_stride = 128
    g1[6] = 0; g1[7] = 0;
    const i32x4 z4 = {};
    const i32x8 z8 = {};
    // 6-arg toolchain form: (g0, g1, g2, g3, extra, cpol)
    __builtin_amdgcn_tensor_load_to_lds(g0, g1, z4, z4, z8, 0);
    __builtin_amdgcn_s_wait_tensorcnt(0);
  }
#else
  for (int idx = tid; idx < IL * TL; idx += 128)
    A_s[(idx >> 7) * 129 + (idx & 127)] = Ag[idx];
#endif
  __syncthreads();

  // T0 = 1 where unmasked (A > 0 exactly where unmasked)
  for (int idx = tid; idx < IL * TL; idx += 128) {
    const int li = (idx >> 7) * 129 + (idx & 127);
    T_s[li] = (A_s[li] > 0.f) ? 1.f : 0.f;
  }
  __syncthreads();

  for (int it = 0; it < NITER; ++it) {
    // delta[n] = 1 / (yl * sum_m Q[n][m]*sigma[m] + ymask[n]),  Q = A.*T
    if (tid < IL) {
      const float* Ar = A_s + tid * 129;
      const float* Tr = T_s + tid * 129;
      float acc = 0.f;
      for (int m = 0; m < TL; ++m) acc += Ar[m] * Tr[m] * sigma_s[m];
      delta_s[tid] = 1.0f / (yl * acc + ymask[tid]);
    }
    __syncthreads();
    // sigma[m] = 1 / (xl * sum_n Q[n][m]*delta[n] + xmask[m])
    {
      float acc = 0.f;
      for (int n = 0; n < IL; ++n) {
        const int li = n * 129 + tid;
        acc += A_s[li] * T_s[li] * delta_s[n];
      }
      sigma_s[tid] = 1.0f / (xl * acc + xmask[tid]);
    }
    __syncthreads();
    // T[n][m] = delta[n] * (A.*T)[n][m] * sigma[m]
    for (int idx = tid; idx < IL * TL; idx += 128) {
      const int n = idx >> 7, m = idx & 127;
      const int li = n * 129 + m;
      T_s[li] = delta_s[n] * (A_s[li] * T_s[li]) * sigma_s[m];
    }
    __syncthreads();
  }

  // sum( C^T .* T ), C^T = -beta*log(A) where A>0 (masked entries have T=0)
  float part = 0.f;
  for (int idx = tid; idx < IL * TL; idx += 128) {
    const int li = (idx >> 7) * 129 + (idx & 127);
    const float a = A_s[li];
    if (a > 0.f) part += (-0.5f * logf(a)) * T_s[li];
  }
  const float tot = block_reduce_sum(part, red);
  if (tid == 0) partials[b] = tot;
}

// ---------------------------------------------------------------------------
// Kernel 4: deterministic fixed-order final reduction -> 0.01 * sum
// ---------------------------------------------------------------------------
__global__ __launch_bounds__(256) void finalize_kernel(
    const float* __restrict__ partials, float* __restrict__ out) {
  __shared__ float s[256];
  const int tid = threadIdx.x;
  s[tid] = partials[tid];
  __syncthreads();
  for (int off = 128; off > 0; off >>= 1) {
    if (tid < off) s[tid] += s[tid + off];
    __syncthreads();
  }
  if (tid == 0) out[0] = 0.01f * s[0];
}

// ---------------------------------------------------------------------------
extern "C" void kernel_launch(void* const* d_in, const int* in_sizes, int n_in,
                              void* d_out, int out_size, void* d_ws, size_t ws_size,
                              hipStream_t stream) {
  (void)in_sizes; (void)n_in; (void)out_size; (void)ws_size;
  const float* txt  = (const float*)d_in[0];   // (256,128,1024) f32
  const float* obj  = (const float*)d_in[1];   // (256,128,1024) f32
  const int*   tnum = (const int*)d_in[2];     // (256,128) i32
  const int*   onum = (const int*)d_in[3];     // (256,128) i32

  float* invx     = (float*)d_ws;                          // 256*128
  float* invy     = invx + BATCH * TL;                     // 256*127
  float* Amat     = invy + BATCH * TL;                     // 256*128*128
  float* partials = Amat + (size_t)BATCH * 128 * 128;      // 256

  rownorm_kernel<<<(BATCH * TL + 7) / 8, 256, 0, stream>>>(txt, invx, TL, 0, TL, BATCH * TL);
  rownorm_kernel<<<(BATCH * IL + 7) / 8, 256, 0, stream>>>(obj, invy, IL, 1, 128, BATCH * IL);

  cosine_gemm_kernel<<<dim3(64, BATCH), 32, 0, stream>>>(
      txt, obj, invx, invy, tnum, onum, Amat);

  const size_t smem = (size_t)(2 * IL * 129 + 5 * 128) * sizeof(float);  // ~131 KB
  ipot_kernel<<<BATCH, 128, smem, stream>>>(Amat, tnum, onum, partials);

  finalize_kernel<<<1, 256, 0, stream>>>(partials, (float*)d_out);
}
